// MoELayer_27410481283326
// MI455X (gfx1250) — compile-verified, matrix operand order checked by reading, not run on hardware
//
#include <hip/hip_runtime.h>
#include <hip/hip_bf16.h>

// ---------------------------------------------------------------------------
// MoE forward for MI455X (gfx1250): bf16 WMMA GEMMs with async global->LDS
// double-buffered staging + VALU router/losses.
// ---------------------------------------------------------------------------

typedef __bf16 bf16_t;
typedef bf16_t v16bf __attribute__((ext_vector_type(16)));
typedef float  v8f   __attribute__((ext_vector_type(8)));

#define NEXP 8
#define TOPK 4
#define NM   16   // 8 real + 8 null slots

// ---- helpers --------------------------------------------------------------

__device__ __forceinline__ unsigned short f2bf(float f) {
  unsigned int u = __float_as_uint(f);
  u += 0x7FFFu + ((u >> 16) & 1u);       // round-to-nearest-even
  return (unsigned short)(u >> 16);
}

__device__ __forceinline__ float gelu_tanh(float x) {
  const float c0 = 0.7978845608028654f;  // sqrt(2/pi)
  float t = tanhf(c0 * (x + 0.044715f * x * x * x));
  return 0.5f * x * (1.0f + t);
}

// gfx1250 async global->LDS copy: ASYNCcnt-tracked, no VGPR staging.
// NOTE: the offset: immediate is added to BOTH the memory and LDS addresses,
// so one (lds,mem) address pair covers a 32-byte strip as two B128 ops.
__device__ __forceinline__ void async_tile_load(unsigned la, unsigned lb,
                                                const unsigned short* pa,
                                                const unsigned short* pb) {
  asm volatile(
      "global_load_async_to_lds_b128 %0, %2, off\n\t"
      "global_load_async_to_lds_b128 %0, %2, off offset:16\n\t"
      "global_load_async_to_lds_b128 %1, %3, off\n\t"
      "global_load_async_to_lds_b128 %1, %3, off offset:16"
      :
      : "v"(la), "v"(lb),
        "v"((unsigned long long)pa), "v"((unsigned long long)pb)
      : "memory");
}

__device__ __forceinline__ void wait_async_zero() {
  asm volatile("s_wait_asynccnt 0x0" ::: "memory");
}

// ---- f32 -> bf16 conversion ------------------------------------------------

__global__ __launch_bounds__(256) void cvt_f32_bf16(const float* __restrict__ in,
                                                    unsigned short* __restrict__ out,
                                                    long long n) {
  long long i = (long long)blockIdx.x * blockDim.x + threadIdx.x;
  long long stride = (long long)gridDim.x * blockDim.x;
  for (; i < n; i += stride) out[i] = f2bf(in[i]);
}

// ---- stats init ------------------------------------------------------------
// stats[0] = sum(lse^2), stats[1..16] = sum softmax P, stats[17..32] = counts
__global__ void init_stats(float* __restrict__ stats) {
  int i = threadIdx.x;
  if (i < 33) stats[i] = 0.0f;
}

// ---- router ----------------------------------------------------------------

__global__ __launch_bounds__(256) void router_kernel(
    const float* __restrict__ x, const float* __restrict__ rw,
    float* __restrict__ gates, float* __restrict__ stats, int nt) {
  const int D = 1024;
  __shared__ float sw[(NEXP + 1) * 1024];   // 36 KB of the 320 KB WGP LDS
  __shared__ float sP[NM];
  __shared__ float sC[NM];
  __shared__ float sZ;

  const int tid = threadIdx.x;
  for (int i = tid; i < (NEXP + 1) * D; i += blockDim.x) sw[i] = rw[i];
  if (tid < NM) { sP[tid] = 0.f; sC[tid] = 0.f; }
  if (tid == 0) sZ = 0.f;
  __syncthreads();

  const int tok = blockIdx.x * blockDim.x + tid;
  if (tok < nt) {
    float s[NEXP + 1];
#pragma unroll
    for (int e = 0; e <= NEXP; ++e) s[e] = 0.f;
    const float4* xr = (const float4*)(x + (size_t)tok * D);
    for (int d4 = 0; d4 < D / 4; ++d4) {
      float4 xv = xr[d4];
#pragma unroll
      for (int e = 0; e <= NEXP; ++e) {
        const float4 wv = *(const float4*)&sw[e * D + d4 * 4];
        s[e] += xv.x * wv.x + xv.y * wv.y + xv.z * wv.z + xv.w * wv.w;
      }
    }
    float ex[NM];
#pragma unroll
    for (int e = 0; e < NEXP; ++e) ex[e] = s[e];
#pragma unroll
    for (int e = NEXP; e < NM; ++e) ex[e] = s[NEXP];

    float mx = ex[0];
#pragma unroll
    for (int i = 1; i < NM; ++i) mx = fmaxf(mx, ex[i]);
    float se = 0.f;
#pragma unroll
    for (int i = 0; i < NM; ++i) se += __expf(ex[i] - mx);
    float lse = mx + __logf(se);
    float inv_se = 1.0f / se;
#pragma unroll
    for (int i = 0; i < NM; ++i) atomicAdd(&sP[i], __expf(ex[i] - mx) * inv_se);
    atomicAdd(&sZ, lse * lse);

    // top-4 with lowest-index tie-break (matches lax.top_k)
    bool used[NM];
#pragma unroll
    for (int i = 0; i < NM; ++i) used[i] = false;
    int ki[TOPK]; float kv[TOPK];
#pragma unroll
    for (int k = 0; k < TOPK; ++k) {
      int best = 0; float bv = -3.4e38f;
#pragma unroll
      for (int i = 0; i < NM; ++i)
        if (!used[i] && ex[i] > bv) { bv = ex[i]; best = i; }
      used[best] = true; ki[k] = best; kv[k] = bv;
      atomicAdd(&sC[best], 1.0f);
    }
    float g[TOPK], gs = 0.f;
#pragma unroll
    for (int k = 0; k < TOPK; ++k) { g[k] = __expf(kv[k] - kv[0]); gs += g[k]; }
    float rsum = 0.f;
#pragma unroll
    for (int k = 0; k < TOPK; ++k) {
      g[k] /= gs;
      if (ki[k] < NEXP) rsum += g[k];
    }
    float gr[NEXP];
#pragma unroll
    for (int e = 0; e < NEXP; ++e) gr[e] = 0.f;
    if (rsum > 0.f) {
      float denom = fmaxf(rsum, 1e-9f);
#pragma unroll
      for (int k = 0; k < TOPK; ++k)
        if (ki[k] < NEXP) gr[ki[k]] = g[k] / denom;
    }
#pragma unroll
    for (int e = 0; e < NEXP; ++e) gates[(size_t)tok * NEXP + e] = gr[e];
  }
  __syncthreads();
  if (tid < NM) {
    atomicAdd(&stats[1 + tid], sP[tid]);
    atomicAdd(&stats[17 + tid], sC[tid]);
  }
  if (tid == 0) atomicAdd(&stats[0], sZ);
}

__global__ void finalize_losses(const float* __restrict__ stats,
                                float* __restrict__ out_losses, int nt) {
  if (threadIdx.x == 0) {
    float invNtK = 1.0f / ((float)nt * (float)TOPK);
    float invNt = 1.0f / (float)nt;
    float bal = 0.f;
    for (int i = 0; i < NM; ++i)
      bal += (stats[17 + i] * invNtK) * (stats[1 + i] * invNt);
    out_losses[0] = (float)NM * bal;    // balance_loss
    out_losses[1] = stats[0] * invNt;   // z_loss = mean(lse^2)
  }
}

// ---- bf16 WMMA GEMM:  C[M,N] = A[M,K] * B[N,K]^T --------------------------
// 256 threads = 8 wave32; block tile 128x128; K-step 32.
// Double-buffered 2x16KB LDS filled with GLOBAL_LOAD_ASYNC_TO_LDS_B128
// (ASYNCcnt), one barrier per K-step, async fill of K+32 overlaps WMMA on K.
// MODE 0: C = bf16, fused tanh-GELU (activation buffers)
// MODE 1: C = f32, plain store       (shared-FFN down: initializes d_out)
// MODE 2: C = f32, += gate[row,expert] * val (expert down, stream-serialized)

template <int MODE>
__global__ __launch_bounds__(256) void gemm_bf16_wmma(
    const unsigned short* __restrict__ A, const unsigned short* __restrict__ B,
    void* __restrict__ Cv, const float* __restrict__ gates,
    int M, int N, int K, int expert) {
  __shared__ __align__(32) unsigned short sA[2][128 * 32];
  __shared__ __align__(32) unsigned short sB[2][128 * 32];

  const int tid  = threadIdx.x;
  const int lane = tid & 31;
  const int wave = tid >> 5;
  const int wm = wave & 3;    // wave M position (4)
  const int wn = wave >> 2;   // wave N position (2)

  const int blockM = blockIdx.y * 128;
  const int blockN = blockIdx.x * 128;

  const v8f vzero = {0.f, 0.f, 0.f, 0.f, 0.f, 0.f, 0.f, 0.f};
  v8f acc[2][4];
#pragma unroll
  for (int i = 0; i < 2; ++i)
#pragma unroll
    for (int j = 0; j < 4; ++j) acc[i][j] = vzero;

  // cooperative tile fill: each thread owns a 16-bf16 (32B) strip
  const int ldr = tid >> 1;          // 0..127 row in tile
  const int ldc = (tid & 1) << 4;    // 0 or 16 (K offset)
  const size_t aBase = (size_t)(blockM + ldr) * K + ldc;
  const size_t bBase = (size_t)(blockN + ldr) * K + ldc;
  const int sOff = ldr * 32 + ldc;

  // hardware LDS byte addresses of this thread's strips (low 32 bits of the
  // shared-aperture flat address are the LDS offset)
  const unsigned lA0 = (unsigned)(unsigned long long)&sA[0][sOff];
  const unsigned lA1 = (unsigned)(unsigned long long)&sA[1][sOff];
  const unsigned lB0 = (unsigned)(unsigned long long)&sB[0][sOff];
  const unsigned lB1 = (unsigned)(unsigned long long)&sB[1][sOff];

  // per-lane fragment addressing within a 16x32 subtile
  const int fr = lane & 15;
  const int fk = (lane >> 4) << 4;

  // prologue: fill buffer 0 with K-step 0
  async_tile_load(lA0, lB0, A + aBase, B + bBase);

  int buf = 0;
  for (int k0 = 0; k0 < K; k0 += 32) {
    wait_async_zero();     // this wave's fills of `buf` are done
    __syncthreads();       // all waves filled `buf`; all done reading `buf^1`
    if (k0 + 32 < K) {     // overlap: fill other buffer while computing
      async_tile_load(buf ? lA0 : lA1, buf ? lB0 : lB1,
                      A + aBase + k0 + 32, B + bBase + k0 + 32);
      if (k0 + 64 < K) {   // global_prefetch_b8 a step further into L2
        __builtin_prefetch(A + aBase + k0 + 64, 0, 1);
        __builtin_prefetch(B + bBase + k0 + 64, 0, 1);
      }
    }
    v16bf af[2], bf[4];
#pragma unroll
    for (int i = 0; i < 2; ++i)
      af[i] = *(const v16bf*)&sA[buf][(wm * 32 + i * 16 + fr) * 32 + fk];
#pragma unroll
    for (int j = 0; j < 4; ++j)
      bf[j] = *(const v16bf*)&sB[buf][(wn * 64 + j * 16 + fr) * 32 + fk];
#pragma unroll
    for (int i = 0; i < 2; ++i)
#pragma unroll
      for (int j = 0; j < 4; ++j)
        acc[i][j] = __builtin_amdgcn_wmma_f32_16x16x32_bf16(
            false, af[i], false, bf[j], (short)0, acc[i][j], false, false);
    buf ^= 1;
  }

  // epilogue: 16x16 f32 C layout — VGPR v: rows v (lanes 0-15) / v+8 (16-31)
#pragma unroll
  for (int i = 0; i < 2; ++i) {
#pragma unroll
    for (int j = 0; j < 4; ++j) {
      const int rBase = blockM + wm * 32 + i * 16 + ((lane >> 4) << 3);
      const int c = blockN + wn * 64 + j * 16 + (lane & 15);
#pragma unroll
      for (int v = 0; v < 8; ++v) {
        const int r = rBase + v;
        float val = acc[i][j][v];
        if (MODE == 0) {
          ((unsigned short*)Cv)[(size_t)r * N + c] = f2bf(gelu_tanh(val));
        } else if (MODE == 1) {
          ((float*)Cv)[(size_t)r * N + c] = val;
        } else {
          float g = gates[(size_t)r * NEXP + expert];
          ((float*)Cv)[(size_t)r * N + c] += g * val;
        }
      }
    }
  }
}

// ---------------------------------------------------------------------------

extern "C" void kernel_launch(void* const* d_in, const int* in_sizes, int n_in,
                              void* d_out, int out_size, void* d_ws, size_t ws_size,
                              hipStream_t stream) {
  (void)n_in; (void)out_size; (void)ws_size;
  const float* x        = (const float*)d_in[0];
  const float* router_w = (const float*)d_in[1];
  const float* w1_s     = (const float*)d_in[2];
  const float* w2_s     = (const float*)d_in[3];
  const float* w1_e     = (const float*)d_in[4];
  const float* w2_e     = (const float*)d_in[5];
  float* out = (float*)d_out;

  const int D  = 1024;
  const int nt = in_sizes[0] / D;               // 8192 tokens
  const int Hs = in_sizes[2] / D;               // 4096
  const int He = in_sizes[4] / (NEXP * D);      // 1024

  // workspace carve-out (256B aligned blocks)
  char* ws = (char*)d_ws;
  size_t off = 0;
  auto alloc = [&](size_t bytes) -> char* {
    char* p = ws + off;
    off += (bytes + 255) & ~(size_t)255;
    return p;
  };
  unsigned short* x_bf   = (unsigned short*)alloc((size_t)nt * D * 2);
  unsigned short* w1s_bf = (unsigned short*)alloc((size_t)Hs * D * 2);
  unsigned short* w2s_bf = (unsigned short*)alloc((size_t)D * Hs * 2);
  unsigned short* w1e_bf = (unsigned short*)alloc((size_t)NEXP * He * D * 2);
  unsigned short* w2e_bf = (unsigned short*)alloc((size_t)NEXP * D * He * 2);
  unsigned short* hs_bf  = (unsigned short*)alloc((size_t)nt * Hs * 2);
  unsigned short* he_bf  = (unsigned short*)alloc((size_t)nt * He * 2);
  float* gates = (float*)alloc((size_t)nt * NEXP * 4);
  float* stats = (float*)alloc(64 * 4);

  init_stats<<<1, 64, 0, stream>>>(stats);

  auto cvt = [&](const float* src, unsigned short* dst, long long n) {
    long long b = (n + 255) / 256;
    int blocks = (int)(b > 4096 ? 4096 : b);
    cvt_f32_bf16<<<blocks, 256, 0, stream>>>(src, dst, n);
  };
  cvt(x,    x_bf,   (long long)nt * D);
  cvt(w1_s, w1s_bf, (long long)Hs * D);
  cvt(w2_s, w2s_bf, (long long)D * Hs);
  cvt(w1_e, w1e_bf, (long long)NEXP * He * D);
  cvt(w2_e, w2e_bf, (long long)NEXP * D * He);

  router_kernel<<<nt / 256, 256, 0, stream>>>(x, router_w, gates, stats, nt);
  finalize_losses<<<1, 1, 0, stream>>>(stats, out + (size_t)nt * D, nt);

  dim3 blk(256);
  // shared FFN: hs = gelu(x @ w1_s^T) ; out = hs @ w2_s^T  (writes all of out)
  gemm_bf16_wmma<0><<<dim3(Hs / 128, nt / 128), blk, 0, stream>>>(
      x_bf, w1s_bf, hs_bf, nullptr, nt, Hs, D, 0);
  gemm_bf16_wmma<1><<<dim3(D / 128, nt / 128), blk, 0, stream>>>(
      hs_bf, w2s_bf, out, nullptr, nt, D, Hs, 0);

  // experts: he = gelu(x @ w1_e[n]^T) ; out += gate[:,n] * (he @ w2_e[n]^T)
  for (int n = 0; n < NEXP; ++n) {
    const unsigned short* w1n = w1e_bf + (size_t)n * He * D;
    const unsigned short* w2n = w2e_bf + (size_t)n * D * He;
    gemm_bf16_wmma<0><<<dim3(He / 128, nt / 128), blk, 0, stream>>>(
        x_bf, w1n, he_bf, nullptr, nt, He, D, 0);
    gemm_bf16_wmma<2><<<dim3(D / 128, nt / 128), blk, 0, stream>>>(
        he_bf, w2n, out, gates, nt, D, He, n);
  }
}
